// ProNoG_Prompt_20057497272458
// MI455X (gfx1250) — compile-verified, matrix operand order checked by reading, not run on hardware
//
#include <hip/hip_runtime.h>

// ---------------------------------------------------------------------------
// ProNoG prompt kernel for MI455X (gfx1250, wave32, WMMA + TDM)
//
// Phases:
//   0) pack_w_kernel : repack f32 weight matrices into bf16 WMMA B-fragment
//                      layout (one 32B record per lane per fragment)
//   1) gate_kernel   : weights = sigmoid(tanh(x@W_sim+b_sim)@w_vec + b_vec)
//   2) scatter_kernel: readout[row] += x[col] * weights[col] (native f32 atomics)
//   3) prompt_kernel : out = x + relu(readout@W1+b1)@W2 + b2
// GEMMs run on v_wmma_f32_16x16x32_bf16; packed weights are staged into LDS
// via the Tensor Data Mover (tensor_load_to_lds + s_wait_tensorcnt).
// ---------------------------------------------------------------------------

typedef __attribute__((ext_vector_type(16))) __bf16 v16bf;
typedef __attribute__((ext_vector_type(8)))  float  v8f;
typedef __attribute__((ext_vector_type(4)))  unsigned int v4u;
typedef __attribute__((ext_vector_type(8)))  int    v8i;
typedef __attribute__((ext_vector_type(4)))  int    v4i;

#define N_NODES_C 100000
#define IN_DIM_C  128
#define HIDDEN_C  64

// ---- WMMA fragment layouts (CDNA5 ISA §7.12.2) ------------------
// A 16x32 bf16: lane L<16 -> row M=L, K = {0..7 in v0..3, 16..23 in v4..7}
//               lane L>=16 -> row M=L-16, K = {8..15, 24..31}
static __device__ __forceinline__ v16bf load_a_frag(const float* __restrict__ rowptr,
                                                    int kt, int lane) {
  const int kbase = kt * 32 + ((lane & 16) ? 8 : 0);
  v16bf a;
#pragma unroll
  for (int i = 0; i < 8; ++i) {
    const int k = kbase + ((i < 4) ? (2 * i) : (16 + 2 * (i - 4)));
    a[2 * i]     = (__bf16)rowptr[k];
    a[2 * i + 1] = (__bf16)rowptr[k + 1];
  }
  return a;
}

// B 32x16 bf16 from row-major W[K x ldn] (pack-time only)
static __device__ __forceinline__ v16bf make_b_frag(const float* __restrict__ W,
                                                    int ldn, int kt, int n0, int lane) {
  const int kbase = kt * 32 + ((lane & 16) ? 8 : 0);
  const int n = n0 + (lane & 15);
  v16bf b;
#pragma unroll
  for (int i = 0; i < 8; ++i) {
    const int k = kbase + ((i < 4) ? (2 * i) : (16 + 2 * (i - 4)));
    b[2 * i]     = (__bf16)W[(size_t)k * ldn + n];
    b[2 * i + 1] = (__bf16)W[(size_t)(k + 1) * ldn + n];
  }
  return b;
}

static __device__ __forceinline__ v8f wmma_bf16(v16bf a, v16bf b, v8f c) {
  return __builtin_amdgcn_wmma_f32_16x16x32_bf16(false, a, false, b,
                                                 (short)0, c, false, false);
}

// ---- Tensor Data Mover: 1-D blob copy global -> LDS ------------------------
// D# per CDNA5 ISA. data_size=4B, tile = tensor = ndwords.
// 6-arg builtin form: (v4u g0, v8i g1, v4i g2, v4i g3, v8i g4, i32 cpol)
static __device__ __forceinline__ void tdm_load_to_lds(const void* gptr,
                                                       unsigned lds_off,
                                                       unsigned ndwords) {
  const unsigned long long ga = (unsigned long long)(uintptr_t)gptr;
  v4u g0;
  g0[0] = 1u;                                              // count=1 (valid, user)
  g0[1] = lds_off;                                         // lds_addr [63:32]
  g0[2] = (unsigned)(ga & 0xFFFFFFFFu);                    // global_addr lo
  g0[3] = (unsigned)((ga >> 32) & 0x01FFFFFFu)             // global_addr hi (57b)
        | (2u << 30);                                      // type=2 ("image")
  v8i g1;
  g1[0] = (int)(2u << 16);                                 // wg_mask=0, data_size=4B
  g1[1] = (int)((ndwords & 0xFFFFu) << 16);                // tensor_dim0[15:0]
  g1[2] = (int)((ndwords >> 16) & 0xFFFFu);                // tensor_dim0[31:16]
  g1[3] = (int)((ndwords & 0xFFFFu) << 16);                // tile_dim0
  g1[4] = 0;                                               // tile_dim1/2 unused
  g1[5] = 0; g1[6] = 0; g1[7] = 0;                         // strides unused (1-D)
  const v4i z4 = {0, 0, 0, 0};                             // groups 2/3 disabled
  const v8i z8 = {0, 0, 0, 0, 0, 0, 0, 0};
  __builtin_amdgcn_tensor_load_to_lds(g0, g1, z4, z4, z8, 0);
}

static __device__ __forceinline__ unsigned lds_offset_of(const void* p) {
  // generic pointers to LDS carry the LDS byte offset in the low 32 bits
  return (unsigned)(uintptr_t)p;
}

// ---------------------------------------------------------------------------
__global__ void zero_kernel(float* __restrict__ p, unsigned long long n) {
  unsigned long long i = (unsigned long long)blockIdx.x * blockDim.x + threadIdx.x;
  if (i < n) p[i] = 0.0f;
}

// Pack W[K x ldn] (f32) into bf16 B-fragment records: out[(kt*n_nt+nt)*32+lane]
__global__ void pack_w_kernel(const float* __restrict__ W, int ldn,
                              int n_kt, int n_nt, v16bf* __restrict__ out) {
  const int f    = (blockIdx.x * blockDim.x + threadIdx.x) >> 5;
  const int lane = threadIdx.x & 31;
  if (f >= n_kt * n_nt) return;
  const int kt = f / n_nt;
  const int nt = f % n_nt;
  out[(size_t)f * 32 + lane] = make_b_frag(W, ldn, kt, nt * 16, lane);
}

// ---------------------------------------------------------------------------
// Gate MLP: one wave handles 16 nodes.
__global__ void gate_kernel(const float* __restrict__ x,
                            const v16bf* __restrict__ packWsim,   // 16 frags
                            const float* __restrict__ b_sim,
                            const float* __restrict__ w_vec,
                            const float* __restrict__ b_vec,
                            float* __restrict__ weights,
                            int n_nodes) {
  __shared__ v16bf lds_w[16 * 32];                       // 16 KB packed W_sim

  const int lane  = threadIdx.x & 31;
  if (threadIdx.x < 32) {                                // wave 0 issues the DMA
    tdm_load_to_lds(packWsim, lds_offset_of(&lds_w[0]), 16 * 32 * 8);
    __builtin_amdgcn_s_wait_tensorcnt(0);
  }
  __syncthreads();                                       // all waves before any exit

  const int gwave = (blockIdx.x * blockDim.x + threadIdx.x) >> 5;
  const int m0    = gwave * 16;
  if (m0 >= n_nodes) return;                             // wave-uniform

  int row = m0 + (lane & 15);
  if (row >= n_nodes) row = n_nodes - 1;
  const float* rowptr = x + (size_t)row * IN_DIM_C;

  v16bf afrag[4];
#pragma unroll
  for (int kt = 0; kt < 4; ++kt) afrag[kt] = load_a_frag(rowptr, kt, lane);

  v8f acc[4];
#pragma unroll
  for (int nt = 0; nt < 4; ++nt) {
    v8f c = {};
#pragma unroll
    for (int kt = 0; kt < 4; ++kt)
      c = wmma_bf16(afrag[kt], lds_w[(kt * 4 + nt) * 32 + lane], c);
    acc[nt] = c;
  }

  float partial[8];
#pragma unroll
  for (int r = 0; r < 8; ++r) partial[r] = 0.0f;
#pragma unroll
  for (int nt = 0; nt < 4; ++nt) {
    const int n = nt * 16 + (lane & 15);
    const float bs = b_sim[n];
    const float wv = w_vec[n];
#pragma unroll
    for (int r = 0; r < 8; ++r)
      partial[r] += tanhf(acc[nt][r] + bs) * wv;
  }

  const float bv = b_vec[0];
#pragma unroll
  for (int r = 0; r < 8; ++r) {
    float s = partial[r];
    s += __shfl_xor(s, 1, 16);
    s += __shfl_xor(s, 2, 16);
    s += __shfl_xor(s, 4, 16);
    s += __shfl_xor(s, 8, 16);
    if ((lane & 15) == 0) {
      const int m = m0 + r + ((lane & 16) ? 8 : 0);
      if (m < n_nodes)
        weights[m] = 1.0f / (1.0f + __expf(-(s + bv)));
    }
  }
}

// ---------------------------------------------------------------------------
// Edge scatter: one wave per edge; each lane moves 4 floats.
__global__ void scatter_kernel(const float* __restrict__ x,
                               const long long* __restrict__ edge_index,
                               const float* __restrict__ weights,
                               float* __restrict__ readout,
                               int n_edges) {
  const int gid  = blockIdx.x * blockDim.x + threadIdx.x;
  const int edge = gid >> 5;
  const int lane = threadIdx.x & 31;
  if (edge >= n_edges) return;

  const long long row = edge_index[edge];                     // edge_index[0][e]
  const long long col = edge_index[(size_t)n_edges + edge];   // edge_index[1][e]
  const float w = weights[col];

  const float4 v = ((const float4*)(x + (size_t)col * IN_DIM_C))[lane];
  float* dst = readout + (size_t)row * IN_DIM_C + lane * 4;
  unsafeAtomicAdd(dst + 0, v.x * w);                          // global_atomic_add_f32
  unsafeAtomicAdd(dst + 1, v.y * w);
  unsafeAtomicAdd(dst + 2, v.z * w);
  unsafeAtomicAdd(dst + 3, v.w * w);
}

// ---------------------------------------------------------------------------
// Prompt MLP: one wave handles 16 nodes.
__global__ void prompt_kernel(const float* __restrict__ x,
                              const float* __restrict__ readout,
                              const v16bf* __restrict__ packW1,   // 16 frags
                              const float* __restrict__ b1,
                              const v16bf* __restrict__ packW2,   // 16 frags
                              const float* __restrict__ b2,
                              float* __restrict__ out,
                              int n_nodes) {
  __shared__ v16bf lds_w1[16 * 32];                      // 16 KB
  __shared__ v16bf lds_w2[16 * 32];                      // 16 KB
  __shared__ float hid[8 * 16 * HIDDEN_C];               // 32 KB (8 wave slices)

  const int lane   = threadIdx.x & 31;
  const int wlocal = threadIdx.x >> 5;
  if (threadIdx.x < 32) {                                // wave 0 issues both DMAs
    tdm_load_to_lds(packW1, lds_offset_of(&lds_w1[0]), 16 * 32 * 8);
    tdm_load_to_lds(packW2, lds_offset_of(&lds_w2[0]), 16 * 32 * 8);
    __builtin_amdgcn_s_wait_tensorcnt(0);
  }
  __syncthreads();                                       // all waves before any exit

  const int gwave = (blockIdx.x * blockDim.x + threadIdx.x) >> 5;
  const int m0    = gwave * 16;
  if (m0 >= n_nodes) return;                             // wave-uniform

  int row = m0 + (lane & 15);
  if (row >= n_nodes) row = n_nodes - 1;
  const float* rowptr = readout + (size_t)row * IN_DIM_C;

  // GEMM1: readout[16x128] @ W1[128x64]
  v16bf afrag[4];
#pragma unroll
  for (int kt = 0; kt < 4; ++kt) afrag[kt] = load_a_frag(rowptr, kt, lane);

  float* hb = &hid[wlocal * 16 * HIDDEN_C];
#pragma unroll
  for (int nt = 0; nt < 4; ++nt) {
    v8f c = {};
#pragma unroll
    for (int kt = 0; kt < 4; ++kt)
      c = wmma_bf16(afrag[kt], lds_w1[(kt * 4 + nt) * 32 + lane], c);
    const int n = nt * 16 + (lane & 15);
    const float bn = b1[n];
    const int moff = (lane & 16) ? 8 : 0;
#pragma unroll
    for (int r = 0; r < 8; ++r)
      hb[(size_t)(r + moff) * HIDDEN_C + n] = fmaxf(c[r] + bn, 0.0f);
  }
  // same-wave LDS RAW: s_wait_dscnt inserted by compiler; no cross-wave sharing

  // GEMM2: hid[16x64] @ W2[64x128]
  const float* hrow = hb + (size_t)(lane & 15) * HIDDEN_C;
  v16bf a2[2];
#pragma unroll
  for (int kt = 0; kt < 2; ++kt) a2[kt] = load_a_frag(hrow, kt, lane);

#pragma unroll
  for (int nt = 0; nt < 8; ++nt) {
    v8f c = {};
#pragma unroll
    for (int kt = 0; kt < 2; ++kt)
      c = wmma_bf16(a2[kt], lds_w2[(kt * 8 + nt) * 32 + lane], c);
    const int n = nt * 16 + (lane & 15);
    const float bn = b2[n];
    const int moff = (lane & 16) ? 8 : 0;
#pragma unroll
    for (int r = 0; r < 8; ++r) {
      const int m = m0 + r + moff;
      if (m < n_nodes) {
        const size_t idx = (size_t)m * IN_DIM_C + n;
        out[idx] = x[idx] + c[r] + bn;
      }
    }
  }
}

// ---------------------------------------------------------------------------
extern "C" void kernel_launch(void* const* d_in, const int* in_sizes, int n_in,
                              void* d_out, int out_size, void* d_ws, size_t ws_size,
                              hipStream_t stream) {
  const float*     x      = (const float*)d_in[0];
  const long long* eidx   = (const long long*)d_in[1];   // int64 [2, E]
  const float*     W_sim  = (const float*)d_in[2];
  const float*     b_sim  = (const float*)d_in[3];
  const float*     w_vec  = (const float*)d_in[4];
  const float*     b_vec  = (const float*)d_in[5];
  const float*     W1     = (const float*)d_in[6];
  const float*     b1     = (const float*)d_in[7];
  const float*     W2     = (const float*)d_in[8];
  const float*     b2     = (const float*)d_in[9];
  float*           out    = (float*)d_out;

  const int n_nodes = N_NODES_C;
  const int n_edges = in_sizes[1] / 2;

  // workspace: readout [N,128] f32 | weights [N] f32 | 3x packed bf16 weights
  float* readout = (float*)d_ws;
  float* weights = readout + (size_t)n_nodes * IN_DIM_C;
  v16bf* packWsim = (v16bf*)(weights + n_nodes);         // 16 KB, 32B aligned
  v16bf* packW1   = packWsim + 16 * 32;
  v16bf* packW2   = packW1   + 16 * 32;

  {
    const unsigned long long n = (unsigned long long)n_nodes * IN_DIM_C;
    zero_kernel<<<(int)((n + 255) / 256), 256, 0, stream>>>(readout, n);
  }
  // pack the three weight matrices into WMMA B-fragment layout (bf16)
  pack_w_kernel<<<2, 256, 0, stream>>>(W_sim, HIDDEN_C, 4, 4, packWsim);
  pack_w_kernel<<<2, 256, 0, stream>>>(W1,    HIDDEN_C, 4, 4, packW1);
  pack_w_kernel<<<2, 256, 0, stream>>>(W2,    IN_DIM_C, 2, 8, packW2);

  {
    const int waves  = (n_nodes + 15) / 16;
    const int blocks = (waves * 32 + 255) / 256;
    gate_kernel<<<blocks, 256, 0, stream>>>(x, packWsim, b_sim, w_vec, b_vec,
                                            weights, n_nodes);
  }
  {
    const long long threads = (long long)n_edges * 32;
    const int blocks = (int)((threads + 255) / 256);
    scatter_kernel<<<blocks, 256, 0, stream>>>(x, eidx, weights, readout, n_edges);
  }
  {
    const int waves  = (n_nodes + 15) / 16;
    const int blocks = (waves * 32 + 255) / 256;
    prompt_kernel<<<blocks, 256, 0, stream>>>(x, readout, packW1, b1, packW2, b2,
                                              out, n_nodes);
  }
}
